// HyperbolicTransformerEmbedding_10728828305787
// MI455X (gfx1250) — matrix operand order repaired and checked by reading, not compiled, not run on hardware
//
#include <hip/hip_runtime.h>
#include <hip/hip_bf16.h>
#include <math.h>

typedef __attribute__((ext_vector_type(16))) _Float16 v16h;
typedef __attribute__((ext_vector_type(8)))  float    v8f;

#define NNODES 20000
#define NEDGES 320000
#define HIDD   64
#define NHEAD  4
#define NLAYER 2
#define INV_SCALE 0.125f   /* 1/sqrt(HID) */

// ---------------------------------------------------------------------------
// wave32 butterfly all-reduce (sum lands in every lane)
// ---------------------------------------------------------------------------
__device__ __forceinline__ float waveAllSum(float v) {
#pragma unroll
  for (int off = 16; off > 0; off >>= 1) v += __shfl_xor(v, off, 32);
  return v;
}

// ---------------------------------------------------------------------------
// Pack f32 weights [J x (K+HT)] -> f16 space-only weights [J x K] (row-aligned)
// ---------------------------------------------------------------------------
__global__ __launch_bounds__(256)
void pack_weights_kernel(const float* __restrict__ src, _Float16* __restrict__ dst,
                         int J, int K, int HT) {
  const int i = blockIdx.x * blockDim.x + threadIdx.x;
  if (i >= J * K) return;
  const int j = i / K, d = i - j * K;
  dst[i] = (_Float16)src[(size_t)j * (K + HT) + HT + d];
}

// ---------------------------------------------------------------------------
// Generic WMMA projection:
//   out[n,j] = sum_d A16[n,d]*W16[j,d] (+ t[n]*W32[j,0]) + bias[j]
//   A16 : [rows x (CHUNKS*32)] f16 row-major
//   W16 : [J x (CHUNKS*32)] f16 row-major (pre-packed, no time column)
//   W32 : original f32 weights [J x (K+HT)] (only col 0 read, when HT=1)
//   One wave computes MT vertically-stacked 16x16 tiles, reusing each
//   B-fragment across MT WMMAs.
//   A-frag layout (ISA 16-bit A 16x32): g = lane>>4; row = lane&15;
//       elems 0..7  -> K = c*32 + g*8 + e ;  elems 8..15 -> +16
//   B-frag layout (ISA 16-bit B 32x16): col = lane&15; elems 0..7 -> K = c*32 + g*8 + e ; 8..15 -> +16
//   D layout: elem r -> (M = g*8 + r, N = lane&15)
// ---------------------------------------------------------------------------
template <int CHUNKS, int HT, int MT>
__global__ __launch_bounds__(128)
void wmma_proj_kernel(const _Float16* __restrict__ A,
                      const _Float16* __restrict__ W16,
                      const float* __restrict__ W32,
                      const float* __restrict__ t,
                      const float* __restrict__ bias,
                      float* __restrict__ out, int J) {
  const int lane    = threadIdx.x & 31;
  const int g       = lane >> 4;
  const int l16     = lane & 15;
  const int rowBase = blockIdx.x * (16 * MT);
  const int jBase   = (blockIdx.y * blockDim.y + threadIdx.y) * 16;
  const int K       = CHUNKS * 32;
  const int j       = jBase + l16;

  v8f acc[MT] = {};
#pragma unroll
  for (int c = 0; c < CHUNKS; ++c) {
    // B fragment: two runs of 8 f16 (16B aligned) at g*8 and g*8+16
    const _Float16* pw = W16 + (size_t)j * K + c * 32 + g * 8;
    v16h b;
#pragma unroll
    for (int e = 0; e < 8; ++e) { b[e] = pw[e]; b[8 + e] = pw[16 + e]; }
#pragma unroll
    for (int r = 0; r < MT; ++r) {
      const _Float16* pa = A + (size_t)(rowBase + r * 16 + l16) * K + c * 32 + g * 8;
      v16h a;
#pragma unroll
      for (int e = 0; e < 8; ++e) { a[e] = pa[e]; a[8 + e] = pa[16 + e]; }
      acc[r] = __builtin_amdgcn_wmma_f32_16x16x32_f16(false, a, false, b,
                                                      (short)0, acc[r], false, false);
    }
  }

  const float w0 = HT ? W32[(size_t)j * (K + 1)] : 0.0f;
  const float bj = bias ? bias[j] : 0.0f;
#pragma unroll
  for (int r = 0; r < MT; ++r) {
#pragma unroll
    for (int m = 0; m < 8; ++m) {
      const int node = rowBase + r * 16 + g * 8 + m;
      float val = acc[r][m] + bj;
      if (HT) val += t[node] * w0;
      out[(size_t)node * J + j] = val;
    }
  }
}

// ---------------------------------------------------------------------------
// expmap0 of raw input: x[N x 128] -> xs16 (f16 space), xtime (f32 time)
// ---------------------------------------------------------------------------
__global__ __launch_bounds__(256)
void expmap_input_kernel(const float* __restrict__ x,
                         _Float16* __restrict__ xs16,
                         float* __restrict__ xtime, int n) {
  const int w    = blockIdx.x * (blockDim.x >> 5) + (threadIdx.x >> 5);
  const int lane = threadIdx.x & 31;
  if (w >= n) return;
  const float* p = x + (size_t)w * 128;
  float v[4];
  float s = 0.0f;
#pragma unroll
  for (int k = 0; k < 4; ++k) { v[k] = p[lane + k * 32]; s += v[k] * v[k]; }
  s = waveAllSum(s);
  const float nr = sqrtf(s + 1e-15f);
  const float sc = sinhf(nr) / nr;
#pragma unroll
  for (int k = 0; k < 4; ++k)
    xs16[(size_t)w * 128 + lane + k * 32] = (_Float16)(v[k] * sc);
  if (lane == 0) xtime[w] = coshf(nr);
}

// ---------------------------------------------------------------------------
// with_time over 64-wide rows: time[r] = sqrt(sum(row^2)+1); optional f16 copy
// ---------------------------------------------------------------------------
__global__ __launch_bounds__(256)
void finalize_h_kernel(const float* __restrict__ hs, float* __restrict__ ht,
                       _Float16* __restrict__ hs16, int rows) {
  const int w    = blockIdx.x * (blockDim.x >> 5) + (threadIdx.x >> 5);
  const int lane = threadIdx.x & 31;
  if (w >= rows) return;
  const float a = hs[(size_t)w * 64 + lane];
  const float b = hs[(size_t)w * 64 + lane + 32];
  if (hs16) {
    hs16[(size_t)w * 64 + lane]      = (_Float16)a;
    hs16[(size_t)w * 64 + lane + 32] = (_Float16)b;
  }
  float s = waveAllSum(a * a + b * b);
  if (lane == 0 && ht) ht[w] = sqrtf(s + 1.0f);
}

// ---------------------------------------------------------------------------
// vtan = logmap0_space(with_time(vs)) computed in place over 64-wide rows
// ---------------------------------------------------------------------------
__global__ __launch_bounds__(256)
void vtan_kernel(float* __restrict__ vs, int rows) {
  const int w    = blockIdx.x * (blockDim.x >> 5) + (threadIdx.x >> 5);
  const int lane = threadIdx.x & 31;
  if (w >= rows) return;
  float a = vs[(size_t)w * 64 + lane];
  float b = vs[(size_t)w * 64 + lane + 32];
  const float s  = waveAllSum(a * a + b * b);
  const float nr = sqrtf(s + 1e-15f);
  const float vt = sqrtf(s + 1.0f);
  const float d  = acoshf(fmaxf(vt, 1.0f + 1e-7f));
  const float sc = d / nr;
  vs[(size_t)w * 64 + lane]      = a * sc;
  vs[(size_t)w * 64 + lane + 32] = b * sc;
}

// ---------------------------------------------------------------------------
// ebias[e,hd] = edge_feats[e,:16] . W_eb[l,hd,:16]
// ---------------------------------------------------------------------------
__global__ __launch_bounds__(256)
void ebias_kernel(const float* __restrict__ ef, const float* __restrict__ Web,
                  float* __restrict__ ebias, int E) {
  const int e = blockIdx.x * blockDim.x + threadIdx.x;
  if (e >= E) return;
  float f[16];
  const float* p = ef + (size_t)e * 16;
#pragma unroll
  for (int i = 0; i < 16; ++i) f[i] = p[i];
#pragma unroll
  for (int hd = 0; hd < NHEAD; ++hd) {
    const float* wv = Web + hd * 16;
    float acc = 0.0f;
#pragma unroll
    for (int i = 0; i < 16; ++i) acc += f[i] * wv[i];
    ebias[(size_t)e * NHEAD + hd] = acc;
  }
}

// ---- monotonic float<->uint key for atomicMax on floats ----
__device__ __forceinline__ unsigned keyEnc(float f) {
  unsigned b = __float_as_uint(f);
  return (b & 0x80000000u) ? ~b : (b | 0x80000000u);
}
__device__ __forceinline__ float keyDec(unsigned k) {
  unsigned b = (k & 0x80000000u) ? (k & 0x7FFFFFFFu) : ~k;
  return __uint_as_float(b);
}

// ---------------------------------------------------------------------------
// Pass 1: per (edge,head) logits + segment max via atomicMax(u32)
// ---------------------------------------------------------------------------
__global__ __launch_bounds__(256)
void edge_logit_max_kernel(const int* __restrict__ src, const int* __restrict__ dst,
                           const float* __restrict__ qs, const float* __restrict__ qt,
                           const float* __restrict__ ks, const float* __restrict__ kt,
                           const float* __restrict__ ebias,
                           float* __restrict__ logits, unsigned* __restrict__ mkey,
                           int E) {
  const int i = blockIdx.x * blockDim.x + threadIdx.x;
  if (i >= E * NHEAD) return;
  const int e = i >> 2, hd = i & 3;
  const int s = src[e], d = dst[e];
  const float* qp = qs + ((size_t)d * NHEAD + hd) * 64;
  const float* kp = ks + ((size_t)s * NHEAD + hd) * 64;
  float inner = -qt[(size_t)d * NHEAD + hd] * kt[(size_t)s * NHEAD + hd];
#pragma unroll 8
  for (int k = 0; k < 64; ++k) inner += qp[k] * kp[k];
  const float logit = (2.0f + 2.0f * inner) * INV_SCALE + ebias[i];
  logits[i] = logit;
  atomicMax(mkey + (size_t)d * NHEAD + hd, keyEnc(logit));
}

// ---------------------------------------------------------------------------
// Pass 2: a = exp(logit - m[dst]); denom[dst] += a  (overwrite logits with a)
// ---------------------------------------------------------------------------
__global__ __launch_bounds__(256)
void edge_exp_sum_kernel(const int* __restrict__ dst,
                         const unsigned* __restrict__ mkey,
                         float* __restrict__ logits, float* __restrict__ denom,
                         int E) {
  const int i = blockIdx.x * blockDim.x + threadIdx.x;
  if (i >= E * NHEAD) return;
  const int e = i >> 2, hd = i & 3;
  const int d = dst[e];
  const float m = keyDec(mkey[(size_t)d * NHEAD + hd]);
  const float a = __expf(logits[i] - m);
  logits[i] = a;
  atomicAdd(denom + (size_t)d * NHEAD + hd, a);
}

// ---------------------------------------------------------------------------
// Pass 3: agg[dst,dim] += sum_hd alpha[e,hd] * vtan[src,hd,dim]  (one wave/edge)
// ---------------------------------------------------------------------------
__global__ __launch_bounds__(256)
void edge_agg_kernel(const int* __restrict__ src, const int* __restrict__ dst,
                     const float* __restrict__ aBuf, const float* __restrict__ denom,
                     const float* __restrict__ vtan, float* __restrict__ agg, int E) {
  const int w    = blockIdx.x * (blockDim.x >> 5) + (threadIdx.x >> 5);
  const int lane = threadIdx.x & 31;
  if (w >= E) return;
  const int s = src[w], d = dst[w];
  float alpha[NHEAD];
#pragma unroll
  for (int hd = 0; hd < NHEAD; ++hd)
    alpha[hd] = aBuf[(size_t)w * NHEAD + hd] / denom[(size_t)d * NHEAD + hd];
  const float* vp = vtan + (size_t)s * (NHEAD * 64);
#pragma unroll
  for (int k = 0; k < 2; ++k) {
    const int dim = lane + k * 32;
    float acc = 0.0f;
#pragma unroll
    for (int hd = 0; hd < NHEAD; ++hd) acc += alpha[hd] * vp[hd * 64 + dim];
    atomicAdd(agg + (size_t)d * 64 + dim, acc);
  }
}

// ---------------------------------------------------------------------------
// Per-node tail: residual merge in tangent space, expmap0, LayerNorm, ReLU,
// with_time.  (logmap0(expmap0(u)) == u used for the aggregated branch.)
// ---------------------------------------------------------------------------
__global__ __launch_bounds__(256)
void node_update_kernel(const float* __restrict__ agg, float* __restrict__ hs,
                        float* __restrict__ ht, _Float16* __restrict__ hs16,
                        const float* __restrict__ lng, const float* __restrict__ lnb,
                        int n) {
  const int w    = blockIdx.x * (blockDim.x >> 5) + (threadIdx.x >> 5);
  const int lane = threadIdx.x & 31;
  if (w >= n) return;
  const float u0 = agg[(size_t)w * 64 + lane]      * (1.0f / NHEAD);
  const float u1 = agg[(size_t)w * 64 + lane + 32] * (1.0f / NHEAD);
  const float r0 = hs[(size_t)w * 64 + lane];
  const float r1 = hs[(size_t)w * 64 + lane + 32];
  const float htv = ht[w];
  const float sr   = waveAllSum(r0 * r0 + r1 * r1);
  const float dres = acoshf(fmaxf(htv, 1.0f + 1e-7f));
  const float fres = dres / sqrtf(sr + 1e-15f);
  const float m0 = 0.5f * (u0 + r0 * fres);
  const float m1 = 0.5f * (u1 + r1 * fres);
  const float sm = waveAllSum(m0 * m0 + m1 * m1);
  const float nm = sqrtf(sm + 1e-15f);
  const float fe = sinhf(nm) / nm;
  float p0 = m0 * fe, p1 = m1 * fe;
  const float mu = waveAllSum(p0 + p1) * (1.0f / 64.0f);
  p0 -= mu; p1 -= mu;
  const float var = waveAllSum(p0 * p0 + p1 * p1) * (1.0f / 64.0f);
  const float inv = rsqrtf(var + 1e-5f);
  float x0 = p0 * inv * lng[lane]      + lnb[lane];
  float x1 = p1 * inv * lng[lane + 32] + lnb[lane + 32];
  x0 = fmaxf(x0, 0.0f); x1 = fmaxf(x1, 0.0f);
  const float st = waveAllSum(x0 * x0 + x1 * x1);
  hs[(size_t)w * 64 + lane]        = x0;
  hs[(size_t)w * 64 + lane + 32]   = x1;
  hs16[(size_t)w * 64 + lane]      = (_Float16)x0;
  hs16[(size_t)w * 64 + lane + 32] = (_Float16)x1;
  if (lane == 0) ht[w] = sqrtf(st + 1.0f);
}

// ---------------------------------------------------------------------------
// Final logmap0_space -> f16 for the output WMMA GEMM
// ---------------------------------------------------------------------------
__global__ __launch_bounds__(256)
void logmap_kernel(const float* __restrict__ hs, const float* __restrict__ ht,
                   _Float16* __restrict__ ltan, int n) {
  const int w    = blockIdx.x * (blockDim.x >> 5) + (threadIdx.x >> 5);
  const int lane = threadIdx.x & 31;
  if (w >= n) return;
  const float a  = hs[(size_t)w * 64 + lane];
  const float b  = hs[(size_t)w * 64 + lane + 32];
  const float s  = waveAllSum(a * a + b * b);
  const float nr = sqrtf(s + 1e-15f);
  const float d  = acoshf(fmaxf(ht[w], 1.0f + 1e-7f));
  const float sc = d / nr;
  ltan[(size_t)w * 64 + lane]      = (_Float16)(a * sc);
  ltan[(size_t)w * 64 + lane + 32] = (_Float16)(b * sc);
}

// ===========================================================================
extern "C" void kernel_launch(void* const* d_in, const int* in_sizes, int n_in,
                              void* d_out, int out_size, void* d_ws, size_t ws_size,
                              hipStream_t stream) {
  (void)in_sizes; (void)n_in; (void)out_size; (void)ws_size;
  const float* x    = (const float*)d_in[0];
  const int*   ei   = (const int*)d_in[1];
  const float* ef   = (const float*)d_in[2];
  const float* W_in = (const float*)d_in[3];
  const float* b_in = (const float*)d_in[4];
  const float* Wq   = (const float*)d_in[5];
  const float* bq   = (const float*)d_in[6];
  const float* Wk   = (const float*)d_in[7];
  const float* bk   = (const float*)d_in[8];
  const float* Wv   = (const float*)d_in[9];
  const float* bv   = (const float*)d_in[10];
  const float* Web  = (const float*)d_in[11];
  const float* lng  = (const float*)d_in[12];
  const float* lnb  = (const float*)d_in[13];
  const float* Wout = (const float*)d_in[14];
  const float* bout = (const float*)d_in[15];
  float* out = (float*)d_out;

  const int N = NNODES, E = NEDGES;
  const int* srcIdx = ei;
  const int* dstIdx = ei + E;

  // ---- workspace carve-out ----
  char* ws = (char*)d_ws;
  size_t off = 0;
  auto carve = [&](size_t bytes) -> char* {
    char* p = ws + off;
    off = (off + bytes + 255) & ~(size_t)255;
    return p;
  };
  _Float16* xs16  = (_Float16*)carve((size_t)N * 128 * 2);
  float*    xtime = (float*)   carve((size_t)N * 4);
  float*    hs    = (float*)   carve((size_t)N * 64 * 4);
  float*    ht    = (float*)   carve((size_t)N * 4);
  _Float16* hs16  = (_Float16*)carve((size_t)N * 64 * 2);
  float*    qs    = (float*)   carve((size_t)N * 256 * 4);
  float*    ks    = (float*)   carve((size_t)N * 256 * 4);
  float*    vs    = (float*)   carve((size_t)N * 256 * 4);   // becomes vtan
  float*    qt    = (float*)   carve((size_t)N * 4 * 4);
  float*    kt    = (float*)   carve((size_t)N * 4 * 4);
  float*    ebias = (float*)   carve((size_t)E * 4 * 4);
  float*    logit = (float*)   carve((size_t)E * 4 * 4);
  unsigned* mkey  = (unsigned*)carve((size_t)N * 4 * 4);
  float*    denom = (float*)   carve((size_t)N * 4 * 4);
  float*    agg   = (float*)   carve((size_t)N * 64 * 4);
  _Float16* ltan  = (_Float16*)carve((size_t)N * 64 * 2);
  // pre-packed f16 weights (space columns only, row-aligned)
  _Float16* Win16  = (_Float16*)carve((size_t)64 * 128 * 2);
  _Float16* Wq16   = (_Float16*)carve((size_t)NLAYER * 256 * 64 * 2);
  _Float16* Wk16   = (_Float16*)carve((size_t)NLAYER * 256 * 64 * 2);
  _Float16* Wv16   = (_Float16*)carve((size_t)NLAYER * 256 * 64 * 2);
  _Float16* Wout16 = (_Float16*)carve((size_t)128 * 64 * 2);

  const dim3 blkRow(256);                 // 8 waves, one row each
  const int  gridN8 = (N + 7) / 8;        // node-row kernels
  const int  gridN4x8 = (N * 4 + 7) / 8;  // (node,head)-row kernels

  // 0) pack all weights to f16 once per launch (tiny)
  pack_weights_kernel<<<(64 * 128 + 255) / 256, blkRow, 0, stream>>>(W_in, Win16, 64, 128, 1);
  pack_weights_kernel<<<(NLAYER * 256 * 64 + 255) / 256, blkRow, 0, stream>>>(Wq, Wq16, NLAYER * 256, 64, 1);
  pack_weights_kernel<<<(NLAYER * 256 * 64 + 255) / 256, blkRow, 0, stream>>>(Wk, Wk16, NLAYER * 256, 64, 1);
  pack_weights_kernel<<<(NLAYER * 256 * 64 + 255) / 256, blkRow, 0, stream>>>(Wv, Wv16, NLAYER * 256, 64, 1);
  pack_weights_kernel<<<(128 * 64 + 255) / 256, blkRow, 0, stream>>>(Wout, Wout16, 128, 64, 0);

  // 1) expmap0 of input -> f16 space + time
  expmap_input_kernel<<<gridN8, blkRow, 0, stream>>>(x, xs16, xtime, N);

  // 2) input projection  (K=128 -> 4 chunks, time column, 2 row tiles/wave)
  {
    dim3 grid(N / 32, 1), blk(32, 4);
    wmma_proj_kernel<4, 1, 2><<<grid, blk, 0, stream>>>(xs16, Win16, W_in, xtime, b_in, hs, 64);
  }
  finalize_h_kernel<<<gridN8, blkRow, 0, stream>>>(hs, ht, hs16, N);

  for (int l = 0; l < NLAYER; ++l) {
    ebias_kernel<<<(E + 255) / 256, blkRow, 0, stream>>>(ef, Web + l * NHEAD * 16, ebias, E);

    // q/k/v projections, all 4 heads fused: [N x 64] x [64 x 256]
    dim3 gridP(N / 32, 4), blkP(32, 4);
    wmma_proj_kernel<2, 1, 2><<<gridP, blkP, 0, stream>>>(
        hs16, Wq16 + (size_t)l * 256 * 64, Wq + (size_t)l * 256 * 65, ht, bq + l * 256, qs, 256);
    wmma_proj_kernel<2, 1, 2><<<gridP, blkP, 0, stream>>>(
        hs16, Wk16 + (size_t)l * 256 * 64, Wk + (size_t)l * 256 * 65, ht, bk + l * 256, ks, 256);
    wmma_proj_kernel<2, 1, 2><<<gridP, blkP, 0, stream>>>(
        hs16, Wv16 + (size_t)l * 256 * 64, Wv + (size_t)l * 256 * 65, ht, bv + l * 256, vs, 256);

    finalize_h_kernel<<<gridN4x8, blkRow, 0, stream>>>(qs, qt, nullptr, N * 4);
    finalize_h_kernel<<<gridN4x8, blkRow, 0, stream>>>(ks, kt, nullptr, N * 4);
    vtan_kernel<<<gridN4x8, blkRow, 0, stream>>>(vs, N * 4);

    hipMemsetAsync(mkey,  0, (size_t)N * 4 * 4, stream);
    hipMemsetAsync(denom, 0, (size_t)N * 4 * 4, stream);
    hipMemsetAsync(agg,   0, (size_t)N * 64 * 4, stream);

    const int eh = E * NHEAD;
    edge_logit_max_kernel<<<(eh + 255) / 256, blkRow, 0, stream>>>(
        srcIdx, dstIdx, qs, qt, ks, kt, ebias, logit, mkey, E);
    edge_exp_sum_kernel<<<(eh + 255) / 256, blkRow, 0, stream>>>(
        dstIdx, mkey, logit, denom, E);
    edge_agg_kernel<<<(E + 7) / 8, blkRow, 0, stream>>>(
        srcIdx, dstIdx, logit, denom, vs, agg, E);

    node_update_kernel<<<gridN8, blkRow, 0, stream>>>(
        agg, hs, ht, hs16, lng + l * 64, lnb + l * 64, N);
  }

  // final: logmap0_space(h) @ W_out.T + b_out
  logmap_kernel<<<gridN8, blkRow, 0, stream>>>(hs, ht, ltan, N);
  {
    dim3 grid(N / 32, 2), blk(32, 4);
    wmma_proj_kernel<2, 0, 2><<<grid, blk, 0, stream>>>(ltan, Wout16, nullptr, nullptr, bout, out, 128);
  }
}
